// AttentionWithEinsum_61074434949346
// MI455X (gfx1250) — compile-verified, matrix operand order checked by reading, not run on hardware
//
#include <hip/hip_runtime.h>

typedef __attribute__((ext_vector_type(16))) _Float16 v16h;
typedef __attribute__((ext_vector_type(8)))  _Float16 v8h;
typedef __attribute__((ext_vector_type(8)))  float    v8f;

#define S_LEN 4096
#define DIM   512
#define NH    8
#define HD    64
// 1/sqrt(64) * log2(e): scores come out in log2 domain so softmax uses exp2
#define QK_SCALE_L2E 0.18033688011112042f
#define NEG_BIG (-1.0e30f)

union V16H { v16h v; v8h h[2]; };

// ---------------------------------------------------------------------------
// Prep 1: convert x [4096][512] f32 -> f16 row-major (vectorized, 8/thread)
// ---------------------------------------------------------------------------
__global__ __launch_bounds__(256)
void cvt_x_kernel(const float* __restrict__ x, _Float16* __restrict__ xh)
{
    const size_t i = ((size_t)blockIdx.x * blockDim.x + threadIdx.x) * 8;
    v8h o;
#pragma unroll
    for (int j = 0; j < 8; ++j) o[j] = (_Float16)x[i + j];
    *(v8h*)(xh + i) = o;
}

// ---------------------------------------------------------------------------
// Prep 2: convert + transpose the four 512x512 f32 weight matrices into
// f16 column-major Wt[col][k] -- the contiguous per-lane WMMA B layout.
// ---------------------------------------------------------------------------
__global__ __launch_bounds__(256)
void wt_kernel(const float* __restrict__ qp, const float* __restrict__ kp,
               const float* __restrict__ vp, const float* __restrict__ pp,
               _Float16* __restrict__ wtq, _Float16* __restrict__ wtk,
               _Float16* __restrict__ wtv, _Float16* __restrict__ wtp)
{
    const float* src;
    _Float16* dst;
    switch (blockIdx.y) {
        case 0:  src = qp; dst = wtq; break;
        case 1:  src = kp; dst = wtk; break;
        case 2:  src = vp; dst = wtv; break;
        default: src = pp; dst = wtp; break;
    }
    const int tid = blockIdx.x * 256 + threadIdx.x;   // 0..262143
    const int c = tid & 511;                          // column (coalesced read)
    const int r = tid >> 9;                           // k-row
    dst[(size_t)c * DIM + r] = (_Float16)src[(size_t)r * DIM + c];
}

// ---------------------------------------------------------------------------
// Kernel 1: QKV projection.  xh[4096][512] f16  x  Wt[512col][512k] f16.
// Q,K stored [head][s][64]; V stored transposed [head*64+d][s].
// One wave computes a 16-row x 64-col (one head) tile, WMMA over K=512.
// Q is pre-scaled by 1/sqrt(HD)*log2(e) so flash softmax can use exp2.
// ---------------------------------------------------------------------------
__global__ __launch_bounds__(128)
void qkv_proj_kernel(const _Float16* __restrict__ xh,
                     const _Float16* __restrict__ wtq,
                     const _Float16* __restrict__ wtk,
                     const _Float16* __restrict__ wtv,
                     _Float16* __restrict__ Qws,
                     _Float16* __restrict__ Kws,
                     _Float16* __restrict__ Vtws)
{
    const int lane = threadIdx.x & 31;
    const int wave = threadIdx.x >> 5;
    const int ln   = lane & 15;
    const int hi   = lane >> 4;                 // lane half (K-striping select)
    const int rowbase = blockIdx.x * 16;
    const int tile = blockIdx.y * 4 + wave;     // 0..23 = {Q,K,V} x 8 heads
    const int mat  = tile >> 3;                 // 0:Q 1:K 2:V
    const int head = tile & 7;
    const int kb_lane = hi * 8;

    const _Float16* wt = (mat == 0) ? wtq : (mat == 1) ? wtk : wtv;

    v8f acc[4] = {};
    const _Float16* ap_base = xh + (size_t)(rowbase + ln) * DIM;
    const _Float16* bp_base = wt + (size_t)(head * HD + ln) * DIM;

    for (int ks = 0; ks < DIM; ks += 32) {
        const _Float16* ap = ap_base + ks + kb_lane;
        V16H a;
        a.h[0] = *(const v8h*)(ap);
        a.h[1] = *(const v8h*)(ap + 16);
#pragma unroll
        for (int ct = 0; ct < 4; ++ct) {
            const _Float16* bp = bp_base + (size_t)(ct * 16) * DIM + ks + kb_lane;
            V16H b;
            b.h[0] = *(const v8h*)(bp);
            b.h[1] = *(const v8h*)(bp + 16);
            acc[ct] = __builtin_amdgcn_wmma_f32_16x16x32_f16(
                false, a.v, false, b.v, (short)0, acc[ct], false, false);
        }
    }

    const float sc = (mat == 0) ? QK_SCALE_L2E : 1.0f;
    if (mat < 2) {
        _Float16* outp = (mat == 0) ? Qws : Kws;
#pragma unroll
        for (int ct = 0; ct < 4; ++ct) {
            const int d = ct * 16 + ln;
#pragma unroll
            for (int v = 0; v < 8; ++v) {
                const int r = rowbase + v + hi * 8;
                outp[((size_t)head * S_LEN + r) * HD + d] = (_Float16)(acc[ct][v] * sc);
            }
        }
    } else {
        // V transposed: lane holds one column d for 8 consecutive rows -> b128 store
#pragma unroll
        for (int ct = 0; ct < 4; ++ct) {
            const int d = ct * 16 + ln;
            v8h pk;
#pragma unroll
            for (int v = 0; v < 8; ++v) pk[v] = (_Float16)acc[ct][v];
            *(v8h*)(Vtws + ((size_t)(head * HD + d)) * S_LEN + rowbase + hi * 8) = pk;
        }
    }
}

// ---------------------------------------------------------------------------
// Kernel 2: flash attention.  One wave = one (head, 16-row Q tile).
// Per 32-key iteration: 4 score WMMAs (K=head dim), online softmax in log2
// domain (max reduced per-iter; denominator kept per-lane and reduced once),
// P transposed C/D->A layout via per-wave LDS, 4 PV WMMAs (K=key dim).
// V operands are loaded before the LDS fence so they fly under the softmax.
// ---------------------------------------------------------------------------
__global__ __launch_bounds__(128)
void flash_attn_kernel(const _Float16* __restrict__ Qws,
                       const _Float16* __restrict__ Kws,
                       const _Float16* __restrict__ Vtws,
                       _Float16* __restrict__ Ows)
{
    __shared__ alignas(16) _Float16 lds[4][16 * 32];
    const int lane = threadIdx.x & 31;
    const int wave = threadIdx.x >> 5;
    const int ln   = lane & 15;
    const int hi   = lane >> 4;
    const int tile = blockIdx.x * 4 + wave;     // 2048 total
    const int head = tile >> 8;                 // 256 q-tiles per head
    const int qbase = (tile & 255) * 16;
    const int kb_lane = hi * 8;

    const _Float16* Qh = Qws  + (size_t)head * S_LEN * HD;
    const _Float16* Kh = Kws  + (size_t)head * S_LEN * HD;
    const _Float16* Vh = Vtws + (size_t)head * HD * S_LEN;
    _Float16* lp = lds[wave];

    // Q A-operands (pre-scaled), head dims 0..31 and 32..63
    V16H qa0, qa1;
    {
        const _Float16* qp = Qh + (size_t)(qbase + ln) * HD + kb_lane;
        qa0.h[0] = *(const v8h*)(qp);
        qa0.h[1] = *(const v8h*)(qp + 16);
        qa1.h[0] = *(const v8h*)(qp + 32);
        qa1.h[1] = *(const v8h*)(qp + 48);
    }

    v8f o[4] = {};
    float mrow[8], lrow[8];    // lrow: per-lane partial denominator
#pragma unroll
    for (int v = 0; v < 8; ++v) { mrow[v] = NEG_BIG; lrow[v] = 0.0f; }

    for (int kb = 0; kb < S_LEN; kb += 32) {
        // ---- scores for keys [kb,kb+16) and [kb+16,kb+32) ----
        v8f s0, s1;
#pragma unroll
        for (int t = 0; t < 2; ++t) {
            const _Float16* kptr = Kh + (size_t)(kb + t * 16 + ln) * HD + kb_lane;
            V16H b0, b1;
            b0.h[0] = *(const v8h*)(kptr);
            b0.h[1] = *(const v8h*)(kptr + 16);
            b1.h[0] = *(const v8h*)(kptr + 32);
            b1.h[1] = *(const v8h*)(kptr + 48);
            v8f st = {};
            st = __builtin_amdgcn_wmma_f32_16x16x32_f16(
                false, qa0.v, false, b0.v, (short)0, st, false, false);
            st = __builtin_amdgcn_wmma_f32_16x16x32_f16(
                false, qa1.v, false, b1.v, (short)0, st, false, false);
            if (t == 0) s0 = st; else s1 = st;
        }

        // ---- V B-operands: independent of scores, issue before the LDS fence
        V16H vb[4];
#pragma unroll
        for (int c = 0; c < 4; ++c) {
            const _Float16* vptr = Vh + (size_t)(c * 16 + ln) * S_LEN + kb + kb_lane;
            vb[c].h[0] = *(const v8h*)(vptr);
            vb[c].h[1] = *(const v8h*)(vptr + 16);
        }

        // ---- online softmax (log2 domain; only the max needs a row reduce)
        float alpha[8];
#pragma unroll
        for (int v = 0; v < 8; ++v) {
            float mx = fmaxf(s0[v], s1[v]);
            mx = fmaxf(mx, __shfl_xor(mx, 1, 32));
            mx = fmaxf(mx, __shfl_xor(mx, 2, 32));
            mx = fmaxf(mx, __shfl_xor(mx, 4, 32));
            mx = fmaxf(mx, __shfl_xor(mx, 8, 32));
            const float mnew = fmaxf(mrow[v], mx);
            alpha[v] = exp2f(mrow[v] - mnew);
            const float p0 = exp2f(s0[v] - mnew);
            const float p1 = exp2f(s1[v] - mnew);
            s0[v] = p0; s1[v] = p1;
            lrow[v] = lrow[v] * alpha[v] + (p0 + p1);   // per-lane partial
            mrow[v] = mnew;
        }
#pragma unroll
        for (int c = 0; c < 4; ++c)
#pragma unroll
            for (int v = 0; v < 8; ++v) o[c][v] *= alpha[v];

        // ---- transpose P (C/D layout -> A layout) via LDS ----
#pragma unroll
        for (int v = 0; v < 8; ++v) {
            const int r = v + hi * 8;
            lp[r * 32 + ln]      = (_Float16)s0[v];
            lp[r * 32 + 16 + ln] = (_Float16)s1[v];
        }
        asm volatile("s_wait_dscnt 0" ::: "memory");
        V16H pa;
        pa.h[0] = *(const v8h*)(lp + ln * 32 + kb_lane);
        pa.h[1] = *(const v8h*)(lp + ln * 32 + 16 + kb_lane);

        // ---- O += P x V ----
#pragma unroll
        for (int c = 0; c < 4; ++c)
            o[c] = __builtin_amdgcn_wmma_f32_16x16x32_f16(
                false, pa.v, false, vb[c].v, (short)0, o[c], false, false);
    }

    // ---- final denominator reduce across the 16-lane row group ----
#pragma unroll
    for (int v = 0; v < 8; ++v) {
        float t = lrow[v];
        t += __shfl_xor(t, 1, 32);
        t += __shfl_xor(t, 2, 32);
        t += __shfl_xor(t, 4, 32);
        t += __shfl_xor(t, 8, 32);
        lrow[v] = 1.0f / t;
    }
    // ---- normalize, store O row-major [s][h*64+d] f16 ----
#pragma unroll
    for (int c = 0; c < 4; ++c) {
        const int d = c * 16 + ln;
#pragma unroll
        for (int v = 0; v < 8; ++v) {
            const int r = qbase + v + hi * 8;
            Ows[(size_t)r * DIM + head * HD + d] = (_Float16)(o[c][v] * lrow[v]);
        }
    }
}

// ---------------------------------------------------------------------------
// Kernel 3: output projection.  O[4096][512] f16  x  Pt[512c][512hd] f16 -> f32.
// ---------------------------------------------------------------------------
__global__ __launch_bounds__(128)
void out_proj_kernel(const _Float16* __restrict__ Ows,
                     const _Float16* __restrict__ wtp,
                     float* __restrict__ out)
{
    const int lane = threadIdx.x & 31;
    const int wave = threadIdx.x >> 5;
    const int ln   = lane & 15;
    const int hi   = lane >> 4;
    const int rowbase = blockIdx.x * 16;
    const int cbase   = (blockIdx.y * 4 + wave) * 64;
    const int kb_lane = hi * 8;

    v8f acc[4] = {};
    const _Float16* ap_base = Ows + (size_t)(rowbase + ln) * DIM;
    const _Float16* bp_base = wtp + (size_t)(cbase + ln) * DIM;

    for (int ks = 0; ks < DIM; ks += 32) {
        const _Float16* ap = ap_base + ks + kb_lane;
        V16H a;
        a.h[0] = *(const v8h*)(ap);
        a.h[1] = *(const v8h*)(ap + 16);
#pragma unroll
        for (int ct = 0; ct < 4; ++ct) {
            const _Float16* bp = bp_base + (size_t)(ct * 16) * DIM + ks + kb_lane;
            V16H b;
            b.h[0] = *(const v8h*)(bp);
            b.h[1] = *(const v8h*)(bp + 16);
            acc[ct] = __builtin_amdgcn_wmma_f32_16x16x32_f16(
                false, a.v, false, b.v, (short)0, acc[ct], false, false);
        }
    }
#pragma unroll
    for (int ct = 0; ct < 4; ++ct) {
        const int c = cbase + ct * 16 + ln;
#pragma unroll
        for (int v = 0; v < 8; ++v) {
            const int r = rowbase + v + hi * 8;
            out[(size_t)r * DIM + c] = acc[ct][v];
        }
    }
}

// ---------------------------------------------------------------------------
extern "C" void kernel_launch(void* const* d_in, const int* in_sizes, int n_in,
                              void* d_out, int out_size, void* d_ws, size_t ws_size,
                              hipStream_t stream) {
    (void)in_sizes; (void)n_in; (void)out_size; (void)ws_size;
    const float* x  = (const float*)d_in[0];
    const float* qp = (const float*)d_in[1];
    const float* kp = (const float*)d_in[2];
    const float* vp = (const float*)d_in[3];
    const float* pp = (const float*)d_in[4];
    float* out = (float*)d_out;

    const size_t mat_elems = (size_t)NH * S_LEN * HD;   // 2M halves
    const size_t w_elems   = (size_t)DIM * DIM;         // 256K halves
    _Float16* Qws = (_Float16*)d_ws;
    _Float16* Kws = Qws + mat_elems;
    _Float16* Vt  = Kws + mat_elems;
    _Float16* Ows = Vt  + mat_elems;
    _Float16* xh  = Ows + mat_elems;                    // 2M halves
    _Float16* wtq = xh  + (size_t)S_LEN * DIM;
    _Float16* wtk = wtq + w_elems;
    _Float16* wtv = wtk + w_elems;
    _Float16* wtp = wtv + w_elems;                      // total ~22 MB of ws

    // One-shot precision/layout prep (L2-resident, bandwidth-trivial)
    cvt_x_kernel<<<dim3((S_LEN * DIM) / (256 * 8)), 256, 0, stream>>>(x, xh);
    wt_kernel<<<dim3((DIM * DIM) / 256, 4), 256, 0, stream>>>(
        qp, kp, vp, pp, wtq, wtk, wtv, wtp);

    // 256 row tiles x 24 wave-tiles (Q/K/V x 8 heads), 4 waves per block
    qkv_proj_kernel<<<dim3(256, 6), 128, 0, stream>>>(xh, wtq, wtk, wtv, Qws, Kws, Vt);
    // 2048 (head, q-tile) waves, 4 waves per block
    flash_attn_kernel<<<dim3(512), 128, 0, stream>>>(Qws, Kws, Vt, Ows);
    // 256 row tiles x 8 column groups, 4 waves per block
    out_proj_kernel<<<dim3(256, 2), 128, 0, stream>>>(Ows, wtp, out);
}
